// TypeWiseHungarianLoss_26706106646869
// MI455X (gfx1250) — compile-verified, matrix operand order checked by reading, not run on hardware
//
#include <hip/hip_runtime.h>
#include <hip/hip_bf16.h>

// Problem constants (match the reference).
#define BB   16
#define NN   1024
#define VV   4096
#define TT   8
#define BN   (BB * NN)      // 16384 rows
#define NGRP (BB * TT)      // 128 (b,type) groups
#define KMAX 112            // Binomial(1024,1/16): mean 64, sd 7.75 -> P(k>112) ~ 0
#define INFF 1e30f

typedef float    v2f  __attribute__((ext_vector_type(2)));
typedef float    v8f  __attribute__((ext_vector_type(8)));
typedef _Float16 v16h __attribute__((ext_vector_type(16)));
typedef int      v4i  __attribute__((ext_vector_type(4)));

typedef __attribute__((address_space(1))) v4i* g4p;  // global int4*
typedef __attribute__((address_space(3))) v4i* l4p;  // LDS int4*

// ---------------------------------------------------------------------------
// Exact fp32 wave(32)-wide sum using the matrix pipe.
// V_WMMA_F32_16X16X4_F32 with A = (x, 0) per lane, B = ones:
//   A[m,K0] = x[lane m], A[m,K2] = x[lane m+16], K1=K3=0  (ISA 16x4 f32 layout)
//   => D[m,n] = x[m] + x[m+16] for every column n.
// Lane L<16 holds rows 0..7 in its 8 D VGPRs, lane L>=16 holds rows 8..15, so
// sum-of-8 + shfl_xor(16) yields the full 32-lane sum in every lane, exactly.
// ---------------------------------------------------------------------------
__device__ __forceinline__ float wave_sum_wmma(float x) {
#if __has_builtin(__builtin_amdgcn_wmma_f32_16x16x4_f32)
  v2f a; a[0] = x;   a[1] = 0.0f;
  v2f b; b[0] = 1.0f; b[1] = 1.0f;
  v8f c = {0.f, 0.f, 0.f, 0.f, 0.f, 0.f, 0.f, 0.f};
  c = __builtin_amdgcn_wmma_f32_16x16x4_f32(false, a, false, b, (short)0, c,
                                            false, false);
  float s = ((c[0] + c[1]) + (c[2] + c[3])) + ((c[4] + c[5]) + (c[6] + c[7]));
  s += __shfl_xor(s, 16, 32);
  return s;
#elif __has_builtin(__builtin_amdgcn_wmma_f32_16x16x32_f16)
  // Same ones-trick via the codegen-confirmed f16 WMMA (partials -> f16).
  v16h a, b;
#pragma unroll
  for (int i = 0; i < 16; ++i) { a[i] = (_Float16)0.0f; b[i] = (_Float16)1.0f; }
  a[0] = (_Float16)x;  // lanes<16: A[m,K0]=x[m]; lanes>=16: A[m,K8]=x[m+16]
  v8f c = {0.f, 0.f, 0.f, 0.f, 0.f, 0.f, 0.f, 0.f};
  c = __builtin_amdgcn_wmma_f32_16x16x32_f16(false, a, false, b, (short)0, c,
                                             false, false);
  float s = ((c[0] + c[1]) + (c[2] + c[3])) + ((c[4] + c[5]) + (c[6] + c[7]));
  s += __shfl_xor(s, 16, 32);
  return s;
#else
#pragma unroll
  for (int off = 16; off; off >>= 1) x += __shfl_xor(x, off, 32);
  return x;
#endif
}

// ---------------------------------------------------------------------------
// Stage one 16KB row of logits into LDS. CDNA5 async-load-to-LDS path when
// the builtin exists (ASYNCcnt + s_wait_asynccnt), else plain float4 copies.
// blockDim.x == 256; 4 x b128 issues per thread-slot cover 4096 floats.
// ---------------------------------------------------------------------------
__device__ __forceinline__ void stage_row(const float* __restrict__ gsrc,
                                          float* __restrict__ lds, int tid) {
#if __has_builtin(__builtin_amdgcn_global_load_async_to_lds_b128)
#pragma unroll
  for (int i = 0; i < 4; ++i) {
    const int e4 = i * 256 + tid;  // float4 index within the row
    __builtin_amdgcn_global_load_async_to_lds_b128(
        (g4p)(gsrc + (size_t)e4 * 4),
        (l4p)(lds + (size_t)e4 * 4),
        0, 0);
  }
#if __has_builtin(__builtin_amdgcn_s_wait_asynccnt)
  __builtin_amdgcn_s_wait_asynccnt(0);
#else
  asm volatile("s_wait_asynccnt 0" ::: "memory");
#endif
#else
#pragma unroll
  for (int i = 0; i < 4; ++i) {
    const int e4 = i * 256 + tid;
    ((float4*)lds)[e4] = ((const float4*)gsrc)[e4];
  }
#endif
  __syncthreads();
}

// ---------------------------------------------------------------------------
// Kernel 1: logZ[row] = max + log(sum(exp(x - max))) for each of 16384 rows.
// One 256-thread workgroup per row; row lives in LDS after one HBM read.
// ---------------------------------------------------------------------------
__global__ __launch_bounds__(256) void logz_kernel(
    const float* __restrict__ logits, float* __restrict__ logZ) {
  __shared__ float row[VV];
  __shared__ float red[8];
  __shared__ float s_rowmax;

  const int tid = threadIdx.x;
  const size_t base = (size_t)blockIdx.x * VV;
  stage_row(logits + base, row, tid);

  // Pass 1: row max.
  float m = -INFF;
#pragma unroll
  for (int i = 0; i < VV / 256; ++i) m = fmaxf(m, row[tid + 256 * i]);
#pragma unroll
  for (int off = 16; off; off >>= 1) m = fmaxf(m, __shfl_xor(m, off, 32));
  const int wave = tid >> 5;
  if ((tid & 31) == 0) red[wave] = m;
  __syncthreads();
  if (tid == 0) {
    float mm = red[0];
#pragma unroll
    for (int w = 1; w < 8; ++w) mm = fmaxf(mm, red[w]);
    s_rowmax = mm;
  }
  __syncthreads();
  const float rowmax = s_rowmax;

  // Pass 2: sum of exp (fp32 partials; WMMA wave reduction; LDS cross-wave).
  float p = 0.0f;
#pragma unroll
  for (int i = 0; i < VV / 256; ++i) p += __expf(row[tid + 256 * i] - rowmax);
  p = wave_sum_wmma(p);  // EXEC is all-ones here (no divergence in this block)
  __syncthreads();
  if ((tid & 31) == 0) red[wave] = p;
  __syncthreads();
  if (tid == 0) {
    float s = 0.0f;
#pragma unroll
    for (int w = 0; w < 8; ++w) s += red[w];
    logZ[blockIdx.x] = rowmax + __logf(s);
  }
}

// ---------------------------------------------------------------------------
// Kernel 2: one workgroup per (b,type) group. Gathers masked positions,
// builds the k x k cost matrix in LDS (-logits; per-row logZ constant cannot
// change the optimum), runs Jonker-Volgenant with the O(k) column scan and
// argmin parallelized across 128 threads, then writes the group's matched
// NLL sum and count to its own workspace slot (deterministic, no atomics).
// ---------------------------------------------------------------------------
__global__ __launch_bounds__(128) void assign_kernel(
    const float* __restrict__ logits, const int* __restrict__ gt,
    const int* __restrict__ types, const unsigned char* __restrict__ mask,
    const float* __restrict__ logZ, float* __restrict__ groupSum,
    float* __restrict__ groupCnt) {
  __shared__ float cost[KMAX][KMAX];
  __shared__ int   idxs[KMAX];
  __shared__ int   gtok[KMAX];
  __shared__ float u[KMAX + 1], vv[KMAX + 1], minv[KMAX + 1];
  __shared__ int   p[KMAX + 1], way[KMAX + 1];
  __shared__ unsigned char used[KMAX + 1];
  __shared__ float redv[128];
  __shared__ int   redi[128];
  __shared__ int   s_k, s_j0, s_i0, s_j1, s_done;
  __shared__ float s_delta;

  const int tid = threadIdx.x;
  const int grp = blockIdx.x;
  const int b = grp / TT;
  const int t = grp % TT;

  if (tid == 0) {
    int cnt = 0;
    for (int n = 0; n < NN; ++n) {
      const int o = b * NN + n;
      if (types[o] == t && mask[o] != 0) {
        if (cnt < KMAX) { idxs[cnt] = n; gtok[cnt] = gt[o]; }
        ++cnt;
      }
    }
    s_k = cnt > KMAX ? KMAX : cnt;
    groupSum[grp] = 0.0f;  // default for empty groups
    groupCnt[grp] = 0.0f;
  }
  __syncthreads();
  const int k = s_k;
  if (k == 0) return;

  // Build cost matrix: cost[r][c] = -logits[b, idx[r], g[c]].
  for (int e = tid; e < k * k; e += 128) {
    const int r = e / k, c = e - r * k;
    cost[r][c] = -logits[((size_t)b * NN + idxs[r]) * VV + gtok[c]];
  }
  for (int j = tid; j <= k; j += 128) {
    u[j] = 0.0f; vv[j] = 0.0f; p[j] = 0; way[j] = 0;
  }
  __syncthreads();

  for (int i = 1; i <= k; ++i) {
    if (tid == 0) { p[0] = i; s_j0 = 0; }
    for (int j = tid; j <= k; j += 128) { minv[j] = INFF; used[j] = 0; }
    __syncthreads();
    for (;;) {
      if (tid == 0) { used[s_j0] = 1; s_i0 = p[s_j0]; }
      __syncthreads();
      const int j0 = s_j0, i0 = s_i0;
      // Parallel relax over free columns.
      for (int j = 1 + tid; j <= k; j += 128) {
        if (!used[j]) {
          const float cur = cost[i0 - 1][j - 1] - u[i0] - vv[j];
          if (cur < minv[j]) { minv[j] = cur; way[j] = j0; }
        }
      }
      __syncthreads();
      // Parallel argmin over free columns.
      float bv = INFF; int bj = 0;
      for (int j = 1 + tid; j <= k; j += 128) {
        if (!used[j] && minv[j] < bv) { bv = minv[j]; bj = j; }
      }
      redv[tid] = bv; redi[tid] = bj;
      __syncthreads();
      for (int off = 64; off; off >>= 1) {
        if (tid < off && redv[tid + off] < redv[tid]) {
          redv[tid] = redv[tid + off]; redi[tid] = redi[tid + off];
        }
        __syncthreads();
      }
      if (tid == 0) { s_j1 = redi[0]; s_delta = redv[0]; }
      __syncthreads();
      const int j1 = s_j1; const float delta = s_delta;
      // Dual update (p[j] distinct over used columns -> race-free).
      for (int j = tid; j <= k; j += 128) {
        if (used[j]) { u[p[j]] += delta; vv[j] -= delta; }
        else          { minv[j] -= delta; }
      }
      __syncthreads();
      if (tid == 0) { s_j0 = j1; s_done = (p[j1] == 0) ? 1 : 0; }
      __syncthreads();
      if (s_done) break;
    }
    // Augmenting path (serial, O(k)).
    if (tid == 0) {
      int j0 = s_j0;
      while (j0) { const int jp = way[j0]; p[j0] = p[jp]; j0 = jp; }
    }
    __syncthreads();
  }

  // Matched NLL for this group: -(logits[b,pos,tok] - logZ[b,pos]).
  float local = 0.0f;
  for (int j = 1 + tid; j <= k; j += 128) {
    const int r = p[j] - 1;
    const int pos = idxs[r];
    const int tok = gtok[j - 1];
    const float lp =
        logits[((size_t)b * NN + pos) * VV + tok] - logZ[b * NN + pos];
    local -= lp;
  }
  redv[tid] = local;
  __syncthreads();
  for (int off = 64; off; off >>= 1) {
    if (tid < off) redv[tid] += redv[tid + off];
    __syncthreads();
  }
  if (tid == 0) {
    groupSum[grp] = redv[0];
    groupCnt[grp] = (float)k;
  }
}

// ---------------------------------------------------------------------------
// Kernel 3: deterministic final reduction over the 128 group slots.
// ---------------------------------------------------------------------------
__global__ __launch_bounds__(128) void finalize_kernel(
    const float* __restrict__ groupSum, const float* __restrict__ groupCnt,
    float* __restrict__ out) {
  __shared__ float rs[128], rc[128];
  const int tid = threadIdx.x;
  rs[tid] = groupSum[tid];
  rc[tid] = groupCnt[tid];
  __syncthreads();
  for (int off = 64; off; off >>= 1) {
    if (tid < off) { rs[tid] += rs[tid + off]; rc[tid] += rc[tid + off]; }
    __syncthreads();
  }
  if (tid == 0) out[0] = rs[0] / fmaxf(rc[0], 1.0f);
}

extern "C" void kernel_launch(void* const* d_in, const int* in_sizes, int n_in,
                              void* d_out, int out_size, void* d_ws,
                              size_t ws_size, hipStream_t stream) {
  (void)in_sizes; (void)n_in; (void)out_size; (void)ws_size;
  const float*         logits = (const float*)d_in[0];
  const int*           gt     = (const int*)d_in[1];
  const int*           types  = (const int*)d_in[2];
  // jax bool array -> 1 byte per element.
  const unsigned char* mask   = (const unsigned char*)d_in[3];
  float* out = (float*)d_out;

  // Workspace: logZ[BN] | groupSum[NGRP] | groupCnt[NGRP]
  float* logZ     = (float*)d_ws;
  float* groupSum = logZ + BN;
  float* groupCnt = groupSum + NGRP;

  logz_kernel<<<BN, 256, 0, stream>>>(logits, logZ);
  assign_kernel<<<NGRP, 128, 0, stream>>>(logits, gt, types, mask, logZ,
                                          groupSum, groupCnt);
  finalize_kernel<<<1, 128, 0, stream>>>(groupSum, groupCnt, out);
}
